// SGRUModel_64398739636546
// MI455X (gfx1250) — matrix operand order repaired and checked by reading, not compiled, not run on hardware
//
#include <hip/hip_runtime.h>

// ---------------------------------------------------------------------------
// ConvGRU (B=16,T=16,C=3,H=W=64,hd=32) + BN3d + MaxPool(1,2,2) + FC(524288->18)
// Recurrent 3x3 convs run as implicit GEMM on V_WMMA_F32_16X16X32_BF16.
// Data layouts:
//   h bf16, r*h bf16:   zero-padded NHWC [b][py=0..65][px=0..65][c=32]
//                       (halo ring stays zero -> unconditional tap loads)
//   h fp32, z, embed[t]: NHWC [b][y][x][c=32]
//   gx (input-conv output, fp32): [t][b][y][x][96]  (z|r|n gates)
//   packed weights: A-fragment order [mtile][tap][lane][16 bf16]
// ---------------------------------------------------------------------------

typedef __attribute__((ext_vector_type(16))) __bf16 v16bf;
typedef __attribute__((ext_vector_type(8)))  float  v8f;

#define HD 32
#define HW 64
#define PW 66                          // padded spatial extent
#define NB 16
#define NT 16
#define PIX_PER_IMG (HW*HW)            // 4096
#define NPIX (NB*PIX_PER_IMG)          // 65536 pixels per timestep
#define NPAD ((size_t)NB*PW*PW*HD)     // padded bf16 buffer elements: 2230272
#define GX_T_STRIDE ((size_t)NB*PIX_PER_IMG*96)
#define EMB_T_STRIDE ((size_t)NB*PIX_PER_IMG*HD)

__device__ __forceinline__ float sigmoidf_(float v) {
  return 1.0f / (1.0f + __expf(-v));
}

// Element offset of (b, y, x) in the padded NHWC buffer; y/x are UNPADDED
// coords and may be -1..64 (maps onto the zero halo).
__device__ __forceinline__ size_t pidx(int b, int y, int x) {
  return ((size_t)((b * PW + (y + 1)) * PW + (x + 1))) * HD;
}

// Accumulate MT 16x16 output tiles of a 3x3 conv over 32 input channels,
// for a 16-pixel row strip starting at (b, y, x0). src is padded NHWC bf16.
template <int MT>
__device__ __forceinline__ void conv_tiles(const __bf16* __restrict__ src,
                                           const __bf16* __restrict__ wpack,
                                           int b, int y, int x0, int lane,
                                           v8f acc[MT]) {
  const int n  = lane & 15;
  const int kh = lane >> 4;     // which K-half this lane holds for B-fragments
  #pragma unroll
  for (int tap = 0; tap < 9; ++tap) {
    const int dy = tap / 3 - 1;
    const int dx = tap % 3 - 1;
    // Unconditional load: halo ring of the padded buffer is zero.
    const v16bf bfrag =
        *(const v16bf*)(src + pidx(b, y + dy, x0 + n + dx) + kh * 16);
    #pragma unroll
    for (int mt = 0; mt < MT; ++mt) {
      const v16bf afrag =
          *(const v16bf*)(wpack + ((size_t)((mt * 9 + tap) * 32 + lane) * 16));
      acc[mt] = __builtin_amdgcn_wmma_f32_16x16x32_bf16(
          false, afrag, false, bfrag, (short)0, acc[mt], false, false);
    }
  }
}

// ------------------------- init / weight packing ---------------------------

__global__ void __launch_bounds__(256) k_init(float* hf32, __bf16* hbf,
                                              __bf16* rhbf, float* sums,
                                              float* out,
                                              const float* __restrict__ fc_b) {
  const size_t i = (size_t)blockIdx.x * 256 + threadIdx.x;
  if (i < NPAD) { hbf[i] = (__bf16)0.0f; rhbf[i] = (__bf16)0.0f; }
  if (i < (size_t)NPIX * HD) hf32[i] = 0.0f;
  if (i < 64)  sums[i] = 0.0f;
  if (i < NB * 18) out[i] = fc_b[i % 18];
}

// Reorder whzr (64x32x3x3) and whn (32x32x3x3) into WMMA A-fragment order,
// with K ordered as tap*32 + ci.
__global__ void __launch_bounds__(256) k_pack(const float* __restrict__ whzr,
                                              const float* __restrict__ whn,
                                              __bf16* __restrict__ wzrp,
                                              __bf16* __restrict__ wnp) {
  const int i = blockIdx.x * 256 + threadIdx.x;
  if (i < 4 * 9 * 32 * 16) {
    const int e = i & 15, lane = (i >> 4) & 31;
    const int tap = (i >> 9) % 9, mt = i / 4608;
    const int m = lane & 15, kh = lane >> 4;
    const int ci = kh * 16 + e;
    const int co = mt * 16 + m;
    wzrp[i] = (__bf16)whzr[(co * HD + ci) * 9 + tap];
  } else if (i < 4 * 9 * 32 * 16 + 2 * 9 * 32 * 16) {
    const int j = i - 4 * 9 * 32 * 16;
    const int e = j & 15, lane = (j >> 4) & 31;
    const int tap = (j >> 9) % 9, mt = j / 4608;
    const int m = lane & 15, kh = lane >> 4;
    const int ci = kh * 16 + e;
    const int co = mt * 16 + m;
    wnp[j] = (__bf16)whn[(co * HD + ci) * 9 + tap];
  }
}

// ----------------------- input conv (all timesteps) ------------------------
// gx[t][b][pix][96] = conv2d(x[b,t], wx) + bx    (Cin=3, direct, LDS weights)
__global__ void __launch_bounds__(256) k_gx(const float* __restrict__ x,
                                            const float* __restrict__ wx,
                                            const float* __restrict__ bx,
                                            float* __restrict__ gx) {
  __shared__ float sw[96 * 27 + 96];
  for (int i = threadIdx.x; i < 96 * 27; i += 256) sw[i] = wx[i];
  for (int i = threadIdx.x; i < 96; i += 256) sw[96 * 27 + i] = bx[i];
  __syncthreads();

  const int p = blockIdx.x * 256 + threadIdx.x;     // 0 .. 1048575
  const int bt = p >> 12;                           // b*16 + t (x memory order)
  const int pix = p & (PIX_PER_IMG - 1);
  const int y = pix >> 6, xx = pix & 63;

  float in[27];
  #pragma unroll
  for (int ci = 0; ci < 3; ++ci) {
    #pragma unroll
    for (int tap = 0; tap < 9; ++tap) {
      const int yy = y + tap / 3 - 1;
      const int xi = xx + tap % 3 - 1;
      float v = 0.0f;
      if ((unsigned)yy < (unsigned)HW && (unsigned)xi < (unsigned)HW)
        v = x[((size_t)(bt * 3 + ci)) * PIX_PER_IMG + yy * HW + xi];
      in[ci * 9 + tap] = v;
    }
  }
  const int b = bt >> 4, t = bt & 15;
  float* outp = gx + ((size_t)(t * NB + b) * PIX_PER_IMG + pix) * 96;
  for (int co = 0; co < 96; ++co) {
    float a = sw[96 * 27 + co];
    #pragma unroll
    for (int k = 0; k < 27; ++k) a += sw[co * 27 + k] * in[k];
    outp[co] = a;
  }
}

// ---------------------- recurrent step: gates z, r -------------------------
// gh = conv(h, whzr); z = sigmoid(xz+hz); r = sigmoid(xr+hr); rh = r*h
__global__ void __launch_bounds__(256) k_gateA(const __bf16* __restrict__ hbf,
                                               const float* __restrict__ hf32,
                                               const float* __restrict__ gx_t,
                                               const __bf16* __restrict__ wzrp,
                                               const float* __restrict__ bhzr,
                                               float* __restrict__ zbuf,
                                               __bf16* __restrict__ rhbf) {
  const int lane = threadIdx.x & 31;
  const int strip = blockIdx.x * 8 + (threadIdx.x >> 5);   // 4096 strips
  const int b = strip >> 8;
  const int rem = strip & 255;
  const int y = rem >> 2;
  const int x0 = (rem & 3) << 4;

  v8f acc[4] = {};
  conv_tiles<4>(hbf, wzrp, b, y, x0, lane, acc);

  const int n = lane & 15, kh = lane >> 4;
  const int xg = x0 + n;
  const size_t pix = (size_t)(b * HW + y) * HW + xg;
  const size_t pad = pidx(b, y, xg);
  const float* gxp = gx_t + pix * 96;
  #pragma unroll
  for (int mt = 0; mt < 4; ++mt) {
    #pragma unroll
    for (int v = 0; v < 8; ++v) {
      const int co = mt * 16 + kh * 8 + v;
      const float val = acc[mt][v] + bhzr[co];
      if (co < HD) {                         // z-gate channel
        zbuf[pix * HD + co] = sigmoidf_(val + gxp[co]);
      } else {                               // r-gate channel
        const int c = co - HD;
        const float r = sigmoidf_(val + gxp[HD + c]);
        rhbf[pad + c] = (__bf16)(r * hf32[pix * HD + c]);
      }
    }
  }
}

// --------------- recurrent step: candidate n + hidden update ---------------
// n = tanh(xn + conv(rh, whn) + bhn); h = selu((1-z)h + z n); embed[t] = h
__global__ void __launch_bounds__(256) k_gateB(const __bf16* __restrict__ rhbf,
                                               float* __restrict__ hf32,
                                               __bf16* __restrict__ hbf,
                                               const float* __restrict__ gx_t,
                                               const __bf16* __restrict__ wnp,
                                               const float* __restrict__ bhn,
                                               const float* __restrict__ zbuf,
                                               float* __restrict__ embed_t) {
  const int lane = threadIdx.x & 31;
  const int strip = blockIdx.x * 8 + (threadIdx.x >> 5);
  const int b = strip >> 8;
  const int rem = strip & 255;
  const int y = rem >> 2;
  const int x0 = (rem & 3) << 4;

  v8f acc[2] = {};
  conv_tiles<2>(rhbf, wnp, b, y, x0, lane, acc);

  const int n = lane & 15, kh = lane >> 4;
  const int xg = x0 + n;
  const size_t pix = (size_t)(b * HW + y) * HW + xg;
  const size_t pad = pidx(b, y, xg);
  const float* gxp = gx_t + pix * 96;
  const float SELU_S = 1.0507009873554805f;
  const float SELU_A = 1.6732632423543772f;
  #pragma unroll
  for (int mt = 0; mt < 2; ++mt) {
    #pragma unroll
    for (int v = 0; v < 8; ++v) {
      const int c = mt * 16 + kh * 8 + v;
      const float cand = tanhf(gxp[64 + c] + acc[mt][v] + bhn[c]);
      const float z = zbuf[pix * HD + c];
      const float ho = hf32[pix * HD + c];
      const float hl = (1.0f - z) * ho + z * cand;
      const float hn = hl > 0.0f ? SELU_S * hl : SELU_S * SELU_A * (__expf(hl) - 1.0f);
      hf32[pix * HD + c] = hn;
      hbf[pad + c]       = (__bf16)hn;
      embed_t[pix * HD + c] = hn;
    }
  }
}

// -------------------------- BN stats / finalize ----------------------------

__global__ void __launch_bounds__(256) k_bnstats(const float* __restrict__ embed,
                                                 float* __restrict__ sums) {
  __shared__ float ss[256], sq[256];
  const size_t N = (size_t)NT * EMB_T_STRIDE;          // 33554432 elems
  const size_t stride = (size_t)gridDim.x * 256;
  float s = 0.0f, q = 0.0f;
  for (size_t i = (size_t)blockIdx.x * 256 + threadIdx.x; i < N; i += stride) {
    const float v = embed[i];
    s += v; q += v * v;
  }
  ss[threadIdx.x] = s; sq[threadIdx.x] = q;
  __syncthreads();
  if (threadIdx.x < HD) {                              // channel = tid % 32
    float S = 0.0f, Q = 0.0f;
    #pragma unroll
    for (int k = 0; k < 8; ++k) { S += ss[threadIdx.x + 32 * k]; Q += sq[threadIdx.x + 32 * k]; }
    atomicAdd(&sums[threadIdx.x], S);
    atomicAdd(&sums[HD + threadIdx.x], Q);
  }
}

__global__ void k_bnfinal(const float* __restrict__ sums,
                          const float* __restrict__ gamma,
                          const float* __restrict__ beta,
                          float* __restrict__ sc) {
  const int c = threadIdx.x;
  if (c < HD) {
    const float cnt = (float)((size_t)NB * NT * PIX_PER_IMG);  // 1048576
    const float mean = sums[c] / cnt;
    const float var = sums[HD + c] / cnt - mean * mean;
    const float inv = rsqrtf(var + 1e-5f);
    const float scale = gamma[c] * inv;
    sc[c] = scale;
    sc[HD + c] = beta[c] - mean * scale;
  }
}

// ------------------------ normalize + maxpool ------------------------------
// pooled[b][f], f = ((c*16 + t)*32 + y2)*32 + x2   (matches reference reshape)
__global__ void __launch_bounds__(256) k_pool(const float* __restrict__ embed,
                                              const float* __restrict__ sc,
                                              float* __restrict__ pooled) {
  const size_t idx = (size_t)blockIdx.x * 256 + threadIdx.x;
  if (idx >= (size_t)NB * 524288) return;
  const int b = (int)(idx >> 19);
  const int f = (int)(idx & 524287);
  const int x2 = f & 31, y2 = (f >> 5) & 31, t = (f >> 10) & 15, c = f >> 14;
  const float scale = sc[c], shift = sc[HD + c];
  float m = -3.4e38f;
  #pragma unroll
  for (int iy = 0; iy < 2; ++iy)
    #pragma unroll
    for (int ix = 0; ix < 2; ++ix) {
      const size_t ei = ((size_t)(t * NB + b) * PIX_PER_IMG +
                         (2 * y2 + iy) * HW + (2 * x2 + ix)) * HD + c;
      const float v = embed[ei] * scale + shift;
      m = fmaxf(m, v);
    }
  pooled[idx] = m;
}

// --------------------------------- FC --------------------------------------
__global__ void __launch_bounds__(256) k_fc(const float* __restrict__ pooled,
                                            const float* __restrict__ fc_w,
                                            float* __restrict__ out) {
  __shared__ float red[256];
  const int chunk = blockIdx.x, o = blockIdx.y, b = blockIdx.z;
  const float* pb = pooled + (size_t)b * 524288 + (size_t)chunk * 8192;
  const float* wo = fc_w + (size_t)o * 524288 + (size_t)chunk * 8192;
  float acc = 0.0f;
  for (int i = threadIdx.x; i < 8192; i += 256) acc += pb[i] * wo[i];
  red[threadIdx.x] = acc;
  __syncthreads();
  for (int s = 128; s > 0; s >>= 1) {
    if (threadIdx.x < s) red[threadIdx.x] += red[threadIdx.x + s];
    __syncthreads();
  }
  if (threadIdx.x == 0) atomicAdd(&out[b * 18 + o], red[0]);
}

// ---------------------------------------------------------------------------

extern "C" void kernel_launch(void* const* d_in, const int* in_sizes, int n_in,
                              void* d_out, int out_size, void* d_ws, size_t ws_size,
                              hipStream_t stream) {
  (void)in_sizes; (void)n_in; (void)out_size; (void)ws_size;
  const float* x     = (const float*)d_in[0];
  const float* wx    = (const float*)d_in[1];
  const float* bx    = (const float*)d_in[2];
  const float* whzr  = (const float*)d_in[3];
  const float* bhzr  = (const float*)d_in[4];
  const float* whn   = (const float*)d_in[5];
  const float* bhn   = (const float*)d_in[6];
  const float* gamma = (const float*)d_in[7];
  const float* beta  = (const float*)d_in[8];
  const float* fc_w  = (const float*)d_in[9];
  const float* fc_b  = (const float*)d_in[10];
  float* out = (float*)d_out;

  char* w = (char*)d_ws;
  auto take = [&](size_t bytes) -> char* {
    char* p = w;
    w += (bytes + 255) & ~(size_t)255;
    return p;
  };
  float*  hf32   = (float*)take((size_t)NPIX * HD * 4);
  __bf16* hbf    = (__bf16*)take(NPAD * 2);            // padded, halo = 0
  __bf16* rhbf   = (__bf16*)take(NPAD * 2);            // padded, halo = 0
  float*  zbuf   = (float*)take((size_t)NPIX * HD * 4);
  __bf16* wzrp   = (__bf16*)take((size_t)4 * 9 * 32 * 16 * 2);
  __bf16* wnp    = (__bf16*)take((size_t)2 * 9 * 32 * 16 * 2);
  float*  sums   = (float*)take(64 * 4);
  float*  sc     = (float*)take(64 * 4);
  float*  gx     = (float*)take((size_t)NT * GX_T_STRIDE * 4);
  float*  embed  = (float*)take((size_t)NT * EMB_T_STRIDE * 4);
  float*  pooled = (float*)take((size_t)NB * 524288 * 4);

  k_init<<<(int)((NPAD + 255) / 256), 256, 0, stream>>>(hf32, hbf, rhbf, sums,
                                                        out, fc_b);
  k_pack<<<108, 256, 0, stream>>>(whzr, whn, wzrp, wnp);
  k_gx<<<(NB * NT * PIX_PER_IMG) / 256, 256, 0, stream>>>(x, wx, bx, gx);

  for (int t = 0; t < NT; ++t) {
    const float* gx_t = gx + (size_t)t * GX_T_STRIDE;
    float* embed_t = embed + (size_t)t * EMB_T_STRIDE;
    // 4096 strips of 16 pixels, 8 waves per block -> 512 blocks
    k_gateA<<<512, 256, 0, stream>>>(hbf, hf32, gx_t, wzrp, bhzr, zbuf, rhbf);
    k_gateB<<<512, 256, 0, stream>>>(rhbf, hf32, hbf, gx_t, wnp, bhn, zbuf, embed_t);
  }

  k_bnstats<<<2048, 256, 0, stream>>>(embed, sums);
  k_bnfinal<<<1, 32, 0, stream>>>(sums, gamma, beta, sc);
  k_pool<<<(NB * 524288) / 256, 256, 0, stream>>>(embed, sc, pooled);
  k_fc<<<dim3(64, 18, 16), 256, 0, stream>>>(pooled, fc_w, out);
}
